// MultiPropertyMolecularMind_90993177133469
// MI455X (gfx1250) — matrix operand order repaired and checked by reading, not compiled, not run on hardware
//
#include <hip/hip_runtime.h>
#include <hip/hip_bf16.h>

typedef __attribute__((ext_vector_type(16))) _Float16 v16h;
typedef __attribute__((ext_vector_type(8)))  float    v8f;

// ---------------- constants (from setup_inputs) ----------------
#define NB   1024        // molecules
#define NA   32          // atoms per molecule
#define NN   32768       // atoms total
#define NF   64          // atom features
#define NM   200         // molecular features
#define NH   128         // hidden
#define NE   524288      // edges
#define NL   3           // mp layers
#define NP   4           // properties
#define K1P  288         // 259 padded to multiple of 32
#define EPB  64          // edges per block in edge MLP (4 WMMA m-tiles)

// ---------------- optional gfx1250 async global->LDS path ----------------
#if defined(__has_builtin)
# if __has_builtin(__builtin_amdgcn_global_load_async_to_lds_b128) && \
     __has_builtin(__builtin_amdgcn_s_wait_asynccnt)
#  define HAVE_ASYNC_LDS 1
# endif
#endif
#ifndef HAVE_ASYNC_LDS
# define HAVE_ASYNC_LDS 0
#endif

#define AS1 __attribute__((address_space(1)))
#define AS3 __attribute__((address_space(3)))
typedef int async_v4i __attribute__((vector_size(16)));

// ---------------- WMMA fragment helpers (wave32, 16x16x32 f16) ----------------
// A 16x32 (MxK) per ISA 7.12.2: lane<16 -> M=lane, K = {k0..k0+7, k0+16..k0+23}
//                               lane>=16 -> M=lane-16, K shifted +8
__device__ __forceinline__ v16h frag_a_lds(const _Float16* base, int ld, int k0, int lane) {
  int row  = lane & 15;
  int koff = (lane >> 4) << 3;
  const _Float16* p = base + row * ld + k0 + koff;
  v16h a;
#pragma unroll
  for (int j = 0; j < 16; ++j) {
    int k = (j & 7) + ((j >> 3) << 4);
    a[j] = p[k];
  }
  return a;
}

__device__ __forceinline__ v16h frag_a_g32(const float* base, int ld, int row0, int k0, int lane) {
  int row  = row0 + (lane & 15);
  int koff = (lane >> 4) << 3;
  const float* p = base + (size_t)row * ld + k0 + koff;
  v16h a;
#pragma unroll
  for (int j = 0; j < 16; ++j) {
    int k = (j & 7) + ((j >> 3) << 4);
    a[j] = (_Float16)p[k];
  }
  return a;
}

// B 32x16 (KxN), weights stored transposed: Bt[n*ldk + k] -> K = {k0..k0+7, k0+16..k0+23} (+8 for lane>=16), same as A
__device__ __forceinline__ v16h frag_b_t(const _Float16* Bt, int ldk, int k0, int n0, int lane) {
  int col  = n0 + (lane & 15);
  int koff = (lane >> 4) << 3;
  const _Float16* p = Bt + (size_t)col * ldk + k0 + koff;
  v16h b;
#pragma unroll
  for (int j = 0; j < 16; ++j) b[j] = p[(j & 7) + ((j >> 3) << 4)];
  return b;
}

__device__ __forceinline__ v8f wmma16(v16h a, v16h b, v8f c) {
  return __builtin_amdgcn_wmma_f32_16x16x32_f16(false, a, false, b, (short)0, c, false, false);
}

__device__ __forceinline__ float wred_sum(float v) {
#pragma unroll
  for (int off = 16; off > 0; off >>= 1) v += __shfl_xor(v, off, 32);
  return v;
}

// ---------------- utility kernels ----------------
__global__ void k_zero(float* __restrict__ p, int n) {
  int i = blockIdx.x * blockDim.x + threadIdx.x;
  if (i < n) p[i] = 0.f;
}

__global__ void k_cvt_transpose(const float* __restrict__ W, _Float16* __restrict__ o,
                                int K, int N, int Kpad) {
  int i = blockIdx.x * blockDim.x + threadIdx.x;
  if (i >= N * Kpad) return;
  int n = i / Kpad, k = i - n * Kpad;
  o[i] = (k < K) ? (_Float16)W[(size_t)k * N + n] : (_Float16)0.f;
}

__global__ void k_cvt_copy(const float* __restrict__ W, _Float16* __restrict__ o, int n) {
  int i = blockIdx.x * blockDim.x + threadIdx.x;
  if (i < n) o[i] = (_Float16)W[i];
}

__global__ void k_degree(const int* __restrict__ dst, float* __restrict__ cnt, int E) {
  int e = blockIdx.x * blockDim.x + threadIdx.x;
  if (e < E) unsafeAtomicAdd(&cnt[dst[e]], 1.0f);
}

__global__ void k_dist(const float* __restrict__ pos, const int* __restrict__ src,
                       const int* __restrict__ dst, float* __restrict__ distf, int E) {
  int e = blockIdx.x * blockDim.x + threadIdx.x;
  if (e >= E) return;
  int s = src[e], d = dst[e];
  float dx = pos[d * 3 + 0] - pos[s * 3 + 0];
  float dy = pos[d * 3 + 1] - pos[s * 3 + 1];
  float dz = pos[d * 3 + 2] - pos[s * 3 + 2];
  float dd = sqrtf(dx * dx + dy * dy + dz * dz + 1e-12f);
  size_t o = (size_t)e * 3;
  distf[o + 0] = dd;
  distf[o + 1] = 1.f / (1.f + dd);
  distf[o + 2] = expf(-dd);
}

// ---------------- embedding: x = x_atoms @ emb_W + emb_b ----------------
__global__ void k_embed(const float* __restrict__ xa, const _Float16* __restrict__ Wt,
                        const float* __restrict__ bias, float* __restrict__ xf,
                        _Float16* __restrict__ xh) {
  int m0 = blockIdx.x * 16;
  int w = threadIdx.x >> 5, lane = threadIdx.x & 31;
  int n0 = w * 32;
  v8f acc0 = {}, acc1 = {};
#pragma unroll
  for (int kc = 0; kc < 2; ++kc) {
    v16h a  = frag_a_g32(xa, NF, m0, kc * 32, lane);
    v16h b0 = frag_b_t(Wt, NF, kc * 32, n0, lane);
    v16h b1 = frag_b_t(Wt, NF, kc * 32, n0 + 16, lane);
    acc0 = wmma16(a, b0, acc0);
    acc1 = wmma16(a, b1, acc1);
  }
  int col = lane & 15, mb = (lane >> 4) << 3;
#pragma unroll
  for (int r = 0; r < 8; ++r) {
    size_t row = (size_t)(m0 + mb + r) * NH;
    int c0 = n0 + col, c1 = n0 + 16 + col;
    float v0 = acc0[r] + bias[c0];
    float v1 = acc1[r] + bias[c1];
    xf[row + c0] = v0; xh[row + c0] = (_Float16)v0;
    xf[row + c1] = v1; xh[row + c1] = (_Float16)v1;
  }
}

// ---------------- edge MLP + scatter: msg = relu([x_i,x_j,dist]@W1+b1)@W2+b2 ----------------
// 64 edges per block (4 m-tiles) so each B fragment feeds 4 WMMAs.
__global__ void k_edge_mlp(const _Float16* __restrict__ xh, const float* __restrict__ distf,
                           const int* __restrict__ src, const int* __restrict__ dst,
                           const _Float16* __restrict__ W1t, const float* __restrict__ b1,
                           const _Float16* __restrict__ W2t, const float* __restrict__ b2,
                           float* __restrict__ agg) {
  __shared__ __align__(16) _Float16 sA[EPB * K1P];  // 36 KB
  __shared__ __align__(16) _Float16 sH[EPB * NH];   // 16 KB
  __shared__ int sSrc[EPB], sDst[EPB];
  int t = threadIdx.x;
  int e0 = blockIdx.x * EPB;
  if (t < EPB) { sSrc[t] = src[e0 + t]; sDst[t] = dst[e0 + t]; }
  __syncthreads();
  // gather region (cols 0..255): 16-byte chunks, 32 chunks/row
  for (int idx = t; idx < EPB * 32; idx += 128) {
    int row = idx >> 5, ch = idx & 31;
    const _Float16* sp = (ch < 16) ? (xh + (size_t)sDst[row] * NH + ch * 8)
                                   : (xh + (size_t)sSrc[row] * NH + (ch - 16) * 8);
    _Float16* dp = sA + row * K1P + ch * 8;
#if HAVE_ASYNC_LDS
    __builtin_amdgcn_global_load_async_to_lds_b128(
        (AS1 async_v4i*)(unsigned long long)(uintptr_t)sp,
        (AS3 async_v4i*)(unsigned int)(uintptr_t)dp, 0, 0);
#else
    *(float4*)dp = *(const float4*)sp;
#endif
  }
  // dist + zero pad (cols 256..287)
  for (int i = t; i < EPB * 32; i += 128) {
    int row = i >> 5, c = 256 + (i & 31);
    sA[row * K1P + c] = (c < 259) ? (_Float16)distf[(size_t)(e0 + row) * 3 + (c - 256)]
                                  : (_Float16)0.f;
  }
#if HAVE_ASYNC_LDS
  __builtin_amdgcn_s_wait_asynccnt(0);
#endif
  __syncthreads();
  int w = t >> 5, lane = t & 31;
  int n0 = w * 32;
  int col = lane & 15, mb = (lane >> 4) << 3;
  // ---- GEMM1: [64,288] @ [288,128] -> relu -> sH (f16)
  {
    v8f acc[4][2];
#pragma unroll
    for (int mt = 0; mt < 4; ++mt) { acc[mt][0] = (v8f){}; acc[mt][1] = (v8f){}; }
#pragma unroll
    for (int kc = 0; kc < 9; ++kc) {
      if (kc + 1 < 9)
        __builtin_prefetch(W1t + (size_t)(n0 + col) * K1P + (kc + 1) * 32, 0, 3);
      v16h b0 = frag_b_t(W1t, K1P, kc * 32, n0, lane);
      v16h b1 = frag_b_t(W1t, K1P, kc * 32, n0 + 16, lane);
#pragma unroll
      for (int mt = 0; mt < 4; ++mt) {
        v16h a = frag_a_lds(sA + mt * 16 * K1P, K1P, kc * 32, lane);
        acc[mt][0] = wmma16(a, b0, acc[mt][0]);
        acc[mt][1] = wmma16(a, b1, acc[mt][1]);
      }
    }
#pragma unroll
    for (int mt = 0; mt < 4; ++mt) {
#pragma unroll
      for (int r = 0; r < 8; ++r) {
        int c0 = n0 + col, c1 = n0 + 16 + col;
        float v0 = acc[mt][0][r] + b1[c0]; v0 = v0 > 0.f ? v0 : 0.f;
        float v1 = acc[mt][1][r] + b1[c1]; v1 = v1 > 0.f ? v1 : 0.f;
        int rr = mt * 16 + mb + r;
        sH[rr * NH + c0] = (_Float16)v0;
        sH[rr * NH + c1] = (_Float16)v1;
      }
    }
  }
  __syncthreads();
  // ---- GEMM2: [64,128] @ [128,128] -> + b2 -> atomic scatter by dst
  {
    v8f acc[4][2];
#pragma unroll
    for (int mt = 0; mt < 4; ++mt) { acc[mt][0] = (v8f){}; acc[mt][1] = (v8f){}; }
#pragma unroll
    for (int kc = 0; kc < 4; ++kc) {
      v16h b0 = frag_b_t(W2t, NH, kc * 32, n0, lane);
      v16h b1 = frag_b_t(W2t, NH, kc * 32, n0 + 16, lane);
#pragma unroll
      for (int mt = 0; mt < 4; ++mt) {
        v16h a = frag_a_lds(sH + mt * 16 * NH, NH, kc * 32, lane);
        acc[mt][0] = wmma16(a, b0, acc[mt][0]);
        acc[mt][1] = wmma16(a, b1, acc[mt][1]);
      }
    }
    int c0 = n0 + col, c1 = n0 + 16 + col;
    float bb0 = b2[c0], bb1 = b2[c1];
#pragma unroll
    for (int mt = 0; mt < 4; ++mt) {
#pragma unroll
      for (int r = 0; r < 8; ++r) {
        int drow = sDst[mt * 16 + mb + r];
        unsafeAtomicAdd(&agg[(size_t)drow * NH + c0], acc[mt][0][r] + bb0);
        unsafeAtomicAdd(&agg[(size_t)drow * NH + c1], acc[mt][1][r] + bb1);
      }
    }
  }
}

// ---------------- mean-agg + residual + layernorm ----------------
__global__ void k_ln(float* __restrict__ xf, _Float16* __restrict__ xh,
                     const float* __restrict__ agg, const float* __restrict__ cnt,
                     const float* __restrict__ g, const float* __restrict__ bta) {
  int w = threadIdx.x >> 5, lane = threadIdx.x & 31;
  int row = blockIdx.x * 4 + w;
  float cn = fmaxf(cnt[row], 1.0f);
  size_t base = (size_t)row * NH;
  float y[4], s = 0.f, ss = 0.f;
#pragma unroll
  for (int i = 0; i < 4; ++i) {
    int c = lane + 32 * i;
    float v = agg[base + c] / cn + xf[base + c];
    y[i] = v; s += v; ss += v * v;
  }
  s = wred_sum(s); ss = wred_sum(ss);
  float mu = s * (1.f / NH);
  float var = ss * (1.f / NH) - mu * mu;
  float rs = rsqrtf(var + 1e-5f);
#pragma unroll
  for (int i = 0; i < 4; ++i) {
    int c = lane + 32 * i;
    float o = (y[i] - mu) * rs * g[c] + bta[c];
    xf[base + c] = o;
    xh[base + c] = (_Float16)o;
  }
}

// ---------------- per-molecule attention + pooling + pool projection ----------------
__global__ void k_attn(const float* __restrict__ xf, const _Float16* __restrict__ xh,
                       const _Float16* __restrict__ Wqkv, const float* __restrict__ bqkv,
                       const _Float16* __restrict__ Wout, const float* __restrict__ bout,
                       const float* __restrict__ poolW, const float* __restrict__ poolB,
                       float* __restrict__ explicitO, float* __restrict__ learnedO) {
  __shared__ __align__(16) _Float16 sX[NA * NH];        // 8 KB
  __shared__ __align__(16) _Float16 sQKV[NA * 3 * NH];  // 24 KB
  __shared__ __align__(16) _Float16 sO[NA * NH];        // 8 KB
  __shared__ float    sAtt[NA * NH];                    // 16 KB
  __shared__ float    sAw[NA], sAw2[NA];
  __shared__ float    sPool[4 * NH];                    // 2 KB
  int bm = blockIdx.x;
  int t = threadIdx.x;
  size_t xbase = (size_t)bm * NA * NH;
  for (int i = t; i < NA * NH; i += 128) sX[i] = xh[xbase + i];
  __syncthreads();
  int w = t >> 5, lane = t & 31;
  // qkv = xm @ attn_in_w^T + b : [32,128]@[128,384]
  for (int i = 0; i < 6; ++i) {
    int n0 = (w * 6 + i) * 16;
#pragma unroll
    for (int mt = 0; mt < 2; ++mt) {
      v8f acc = {};
#pragma unroll
      for (int kc = 0; kc < 4; ++kc) {
        v16h a = frag_a_lds(sX + mt * 16 * NH, NH, kc * 32, lane);
        v16h b = frag_b_t(Wqkv, NH, kc * 32, n0, lane);
        acc = wmma16(a, b, acc);
      }
      int col = n0 + (lane & 15), mb = mt * 16 + ((lane >> 4) << 3);
#pragma unroll
      for (int r = 0; r < 8; ++r)
        sQKV[(mb + r) * 384 + col] = (_Float16)(acc[r] + bqkv[col]);
    }
  }
  __syncthreads();
  // attention per (head, query-row): 128 threads = 4 heads x 32 rows
  {
    int h = t >> 5, qr = t & 31;
    const float scale = 0.1767766952966369f;  // 1/sqrt(32)
    float sc[32];
    float mx = -3.4e38f;
    for (int kk = 0; kk < 32; ++kk) {
      float acc = 0.f;
      for (int d = 0; d < 32; ++d)
        acc += (float)sQKV[qr * 384 + h * 32 + d] * (float)sQKV[kk * 384 + 128 + h * 32 + d];
      acc *= scale;
      sc[kk] = acc; mx = fmaxf(mx, acc);
    }
    float se = 0.f;
    for (int kk = 0; kk < 32; ++kk) { sc[kk] = __expf(sc[kk] - mx); se += sc[kk]; }
    float inv = 1.f / se;
    for (int d = 0; d < 32; ++d) {
      float o = 0.f;
      for (int kk = 0; kk < 32; ++kk)
        o += sc[kk] * (float)sQKV[kk * 384 + 256 + h * 32 + d];
      sO[qr * NH + h * 32 + d] = (_Float16)(o * inv);
    }
  }
  __syncthreads();
  // attended = o @ attn_out_w^T + b : [32,128]@[128,128]
  for (int i = 0; i < 2; ++i) {
    int n0 = (w * 2 + i) * 16;
#pragma unroll
    for (int mt = 0; mt < 2; ++mt) {
      v8f acc = {};
#pragma unroll
      for (int kc = 0; kc < 4; ++kc) {
        v16h a = frag_a_lds(sO + mt * 16 * NH, NH, kc * 32, lane);
        v16h b = frag_b_t(Wout, NH, kc * 32, n0, lane);
        acc = wmma16(a, b, acc);
      }
      int col = n0 + (lane & 15), mb = mt * 16 + ((lane >> 4) << 3);
#pragma unroll
      for (int r = 0; r < 8; ++r)
        sAtt[(mb + r) * NH + col] = acc[r] + bout[col];
    }
  }
  __syncthreads();
  // attention-pool logits: aw = softmax_a( sum_h attended*xm )
  if (t < NA) {
    float acc = 0.f;
    for (int hh = 0; hh < NH; ++hh)
      acc += sAtt[t * NH + hh] * xf[xbase + t * NH + hh];
    sAw[t] = acc;
  }
  __syncthreads();
  if (t < NA) {
    float mx = -3.4e38f;
    for (int a = 0; a < NA; ++a) mx = fmaxf(mx, sAw[a]);
    float se = 0.f;
    for (int a = 0; a < NA; ++a) se += __expf(sAw[a] - mx);
    sAw2[t] = __expf(sAw[t] - mx) / se;
  }
  __syncthreads();
  // pooling per hidden column (wmean, max, mean, std[ddof=1]) + explicit mean of xm
  {
    int hh = t;
    float wm = 0.f, mxv = -3.4e38f, sm = 0.f, sq = 0.f, ex = 0.f;
    for (int a = 0; a < NA; ++a) {
      float v = sAtt[a * NH + hh];
      wm += v * sAw2[a];
      mxv = fmaxf(mxv, v);
      sm += v; sq += v * v;
      ex += xf[xbase + a * NH + hh];
    }
    float mn = sm * (1.f / NA);
    float sd = sqrtf(fmaxf((sq - NA * mn * mn) / (NA - 1), 0.f));
    sPool[0 * NH + hh] = wm;
    sPool[1 * NH + hh] = mxv;
    sPool[2 * NH + hh] = mn;
    sPool[3 * NH + hh] = sd;
    explicitO[(size_t)bm * NH + hh] = ex * (1.f / NA);
  }
  __syncthreads();
  // learned[b, k*16+q] = pooled[k] @ pool_W[k] + pool_b[k]
  if (t < 64) {
    int k = t >> 4, q = t & 15;
    float acc = poolB[k * 16 + q];
    for (int hh = 0; hh < NH; ++hh)
      acc += sPool[k * NH + hh] * poolW[(k * NH + hh) * 16 + q];
    learnedO[(size_t)bm * 64 + k * 16 + q] = acc;
  }
}

// ---------------- per-property gated MLP heads ----------------
__global__ void k_heads(const float* __restrict__ molF, const float* __restrict__ molGate,
                        const float* __restrict__ lrnGate,
                        const float* __restrict__ meW1, const float* __restrict__ meB1,
                        const float* __restrict__ meW2, const float* __restrict__ meB2,
                        const float* __restrict__ leW1, const float* __restrict__ leB1,
                        const float* __restrict__ leW2, const float* __restrict__ leB2,
                        const float* __restrict__ fW1, const float* __restrict__ fB1,
                        const float* __restrict__ fW2, const float* __restrict__ fB2,
                        const float* __restrict__ hW1, const float* __restrict__ hB1,
                        const float* __restrict__ hW2, const float* __restrict__ hB2,
                        const float* __restrict__ explicitI, const float* __restrict__ learnedI,
                        float* __restrict__ out) {
  __shared__ float sGm[NM];
  __shared__ float sGl[64];
  __shared__ float sT[64];
  __shared__ float sComb[256];
  __shared__ float sF1[128];
  __shared__ float sF2[64];
  __shared__ float sH1[32];
  int b = blockIdx.x, p = blockIdx.y;
  int t = threadIdx.x;  // 64 threads
  for (int m = t; m < NM; m += 64)
    sGm[m] = molF[(size_t)b * NM + m] * (1.f / (1.f + __expf(-molGate[p * NM + m])));
  sGl[t] = learnedI[(size_t)b * 64 + t] * (1.f / (1.f + __expf(-lrnGate[p * 64 + t])));
  for (int c = t; c < 128; c += 64) sComb[c] = explicitI[(size_t)b * NH + c];
  __syncthreads();
  {
    float acc = meB1[p * 64 + t];
    for (int m = 0; m < NM; ++m) acc += sGm[m] * meW1[((size_t)p * NM + m) * 64 + t];
    sT[t] = fmaxf(acc, 0.f);
  }
  __syncthreads();
  {
    float acc = meB2[p * 64 + t];
    for (int j = 0; j < 64; ++j) acc += sT[j] * meW2[((size_t)p * 64 + j) * 64 + t];
    sComb[128 + t] = acc;
  }
  __syncthreads();
  {
    float acc = leB1[p * 64 + t];
    for (int j = 0; j < 64; ++j) acc += sGl[j] * leW1[((size_t)p * 64 + j) * 64 + t];
    sT[t] = fmaxf(acc, 0.f);
  }
  __syncthreads();
  {
    float acc = leB2[p * 64 + t];
    for (int j = 0; j < 64; ++j) acc += sT[j] * leW2[((size_t)p * 64 + j) * 64 + t];
    sComb[192 + t] = acc;
  }
  __syncthreads();
  for (int hh = t; hh < 128; hh += 64) {
    float acc = fB1[p * 128 + hh];
    for (int c = 0; c < 256; ++c) acc += sComb[c] * fW1[((size_t)p * 256 + c) * 128 + hh];
    sF1[hh] = fmaxf(acc, 0.f);
  }
  __syncthreads();
  {
    float acc = fB2[p * 64 + t];
    for (int j = 0; j < 128; ++j) acc += sF1[j] * fW2[((size_t)p * 128 + j) * 64 + t];
    sF2[t] = acc;
  }
  __syncthreads();
  if (t < 32) {
    float acc = hB1[p * 32 + t];
    for (int j = 0; j < 64; ++j) acc += sF2[j] * hW1[((size_t)p * 64 + j) * 32 + t];
    sH1[t] = fmaxf(acc, 0.f);
  }
  __syncthreads();
  if (t == 0) {
    float acc = hB2[p];
    for (int j = 0; j < 32; ++j) acc += sH1[j] * hW2[p * 32 + j];
    out[(size_t)p * NB + b] = acc;
  }
}

// ---------------- launch ----------------
extern "C" void kernel_launch(void* const* d_in, const int* in_sizes, int n_in,
                              void* d_out, int out_size, void* d_ws, size_t ws_size,
                              hipStream_t stream) {
  const float* x_atoms    = (const float*)d_in[0];
  const float* pos        = (const float*)d_in[1];
  const float* molF       = (const float*)d_in[3];
  const float* emb_W      = (const float*)d_in[4];
  const float* emb_b      = (const float*)d_in[5];
  const float* mp_W1      = (const float*)d_in[6];
  const float* mp_b1      = (const float*)d_in[7];
  const float* mp_W2      = (const float*)d_in[8];
  const float* mp_b2      = (const float*)d_in[9];
  const float* ln_g       = (const float*)d_in[10];
  const float* ln_b       = (const float*)d_in[11];
  const float* attn_in_w  = (const float*)d_in[12];
  const float* attn_in_b  = (const float*)d_in[13];
  const float* attn_out_w = (const float*)d_in[14];
  const float* attn_out_b = (const float*)d_in[15];
  const float* pool_W     = (const float*)d_in[16];
  const float* pool_b     = (const float*)d_in[17];
  const float* mol_gate   = (const float*)d_in[18];
  const float* lrn_gate   = (const float*)d_in[19];
  const float* meW1 = (const float*)d_in[20];
  const float* meB1 = (const float*)d_in[21];
  const float* meW2 = (const float*)d_in[22];
  const float* meB2 = (const float*)d_in[23];
  const float* leW1 = (const float*)d_in[24];
  const float* leB1 = (const float*)d_in[25];
  const float* leW2 = (const float*)d_in[26];
  const float* leB2 = (const float*)d_in[27];
  const float* fW1  = (const float*)d_in[28];
  const float* fB1  = (const float*)d_in[29];
  const float* fW2  = (const float*)d_in[30];
  const float* fB2  = (const float*)d_in[31];
  const float* hW1  = (const float*)d_in[32];
  const float* hB1  = (const float*)d_in[33];
  const float* hW2  = (const float*)d_in[34];
  const float* hB2  = (const float*)d_in[35];
  const int* edge_index = (const int*)d_in[36];
  const int* src = edge_index;
  const int* dst = edge_index + NE;
  float* out = (float*)d_out;

  char* ws = (char*)d_ws;
  size_t off = 0;
  auto alloc = [&](size_t bytes) -> void* {
    void* p = ws + off;
    off = (off + bytes + 255) & ~(size_t)255;
    return p;
  };
  float*     xf    = (float*)alloc((size_t)NN * NH * 4);
  float*     agg   = (float*)alloc((size_t)NN * NH * 4);
  _Float16*  xh    = (_Float16*)alloc((size_t)NN * NH * 2);
  float*     cnt   = (float*)alloc((size_t)NN * 4);
  float*     distf = (float*)alloc((size_t)NE * 3 * 4);
  _Float16*  embWt = (_Float16*)alloc((size_t)NF * NH * 2);
  _Float16*  W1t   = (_Float16*)alloc((size_t)NL * K1P * NH * 2);
  _Float16*  W2t   = (_Float16*)alloc((size_t)NL * NH * NH * 2);
  _Float16*  Wqkv  = (_Float16*)alloc((size_t)384 * NH * 2);
  _Float16*  Wout  = (_Float16*)alloc((size_t)NH * NH * 2);
  float*     expl  = (float*)alloc((size_t)NB * NH * 4);
  float*     lrn   = (float*)alloc((size_t)NB * 64 * 4);

  // weight conversion (f32 -> f16, N-major K-contiguous)
  k_cvt_transpose<<<(NH * NF + 255) / 256, 256, 0, stream>>>(emb_W, embWt, NF, NH, NF);
  for (int l = 0; l < NL; ++l) {
    k_cvt_transpose<<<(NH * K1P + 255) / 256, 256, 0, stream>>>(
        mp_W1 + (size_t)l * 259 * NH, W1t + (size_t)l * K1P * NH, 259, NH, K1P);
    k_cvt_transpose<<<(NH * NH + 255) / 256, 256, 0, stream>>>(
        mp_W2 + (size_t)l * NH * NH, W2t + (size_t)l * NH * NH, NH, NH, NH);
  }
  // attn weights: row-major [out,in] == transposed-B layout already
  k_cvt_copy<<<(384 * NH + 255) / 256, 256, 0, stream>>>(attn_in_w, Wqkv, 384 * NH);
  k_cvt_copy<<<(NH * NH + 255) / 256, 256, 0, stream>>>(attn_out_w, Wout, NH * NH);

  // embedding
  k_embed<<<NN / 16, 128, 0, stream>>>(x_atoms, embWt, emb_b, xf, xh);

  // degree counts + distance features
  k_zero<<<(NN + 255) / 256, 256, 0, stream>>>(cnt, NN);
  k_degree<<<(NE + 255) / 256, 256, 0, stream>>>(dst, cnt, NE);
  k_dist<<<(NE + 255) / 256, 256, 0, stream>>>(pos, src, dst, distf, NE);

  // message-passing layers
  for (int l = 0; l < NL; ++l) {
    k_zero<<<(NN * NH + 255) / 256, 256, 0, stream>>>(agg, NN * NH);
    k_edge_mlp<<<NE / EPB, 128, 0, stream>>>(xh, distf, src, dst,
                                             W1t + (size_t)l * K1P * NH, mp_b1 + l * NH,
                                             W2t + (size_t)l * NH * NH, mp_b2 + l * NH, agg);
    k_ln<<<NN / 4, 128, 0, stream>>>(xf, xh, agg, cnt, ln_g + l * NH, ln_b + l * NH);
  }

  // attention + pooling
  k_attn<<<NB, 128, 0, stream>>>(xf, xh, Wqkv, attn_in_b, Wout, attn_out_b,
                                 pool_W, pool_b, expl, lrn);

  // per-property heads
  dim3 hgrid(NB, NP);
  k_heads<<<hgrid, 64, 0, stream>>>(molF, mol_gate, lrn_gate,
                                    meW1, meB1, meW2, meB2,
                                    leW1, leB1, leW2, leB2,
                                    fW1, fB1, fW2, fB2,
                                    hW1, hB1, hW2, hB2,
                                    expl, lrn, out);
}
